// DyTransformer_87342454931917
// MI455X (gfx1250) — matrix operand order repaired and checked by reading, not compiled, not run on hardware
//
#include <hip/hip_runtime.h>
#include <hip/hip_bf16.h>
#include <math.h>

#define T_  8
#define N_  50000
#define D_  128
#define E_  800000
#define L_  2
#define SH_ 8
#define TH_ 8
#define FH_ 16

#define ROWS_  80   // rows per workgroup in structural GEMM (50000 = 625*80)
#define NODES_ 4    // nodes per workgroup in temporal kernel (50000 = 12500*4)

typedef __attribute__((ext_vector_type(16))) __bf16 v16bf;
typedef __attribute__((ext_vector_type(8)))  __bf16 v8bf;   // 16 bytes
typedef __attribute__((ext_vector_type(8)))  float  v8f;

__device__ __forceinline__ __bf16 f2bf(float f) {
  union { float f; unsigned u; } x; x.f = f;
  unsigned r = (x.u + 0x7FFFu + ((x.u >> 16) & 1u)) >> 16;  // round-to-nearest-even
  union { unsigned short s; __bf16 b; } y; y.s = (unsigned short)r;
  return y.b;
}

// 16x16 f32 tile = A[16x128] * B[128x16], K swept in four 16x16x32 bf16 WMMAs.
// A is row-major in LDS (lda=136 -> 272B rows, 16B aligned); B is stored
// TRANSPOSED (Bt[j][d]) so both operands load as contiguous ds_load_b128:
//   A frag: lane m=l&15, hi=l>>4: elems 0..7  = A[m][kb+hi*8 .. +8)
//                                 elems 8..15 = A[m][kb+16+hi*8 .. +8)
//   B frag: col n=l&15: elems 0..7  = Bt[col0+n][kb+hi*8 .. +8)
//                       elems 8..15 = Bt[col0+n][kb+16+hi*8 .. +8)
//   C: col n=l&15; VGPR r -> row r (lanes 0-15) / r+8 (lanes 16-31)
__device__ __forceinline__ v8f wmma_tile(const __bf16* A, int lda,
                                         const __bf16* Bt, int ldt,
                                         int col0, int lane, v8f acc) {
  const int m  = lane & 15;
  const int hi = lane >> 4;
  const __bf16* arow = A + m * lda;
  const __bf16* bcol = Bt + (col0 + m) * ldt;
#pragma unroll
  for (int kb = 0; kb < 128; kb += 32) {
    union { v16bf v; v8bf h[2]; } a, b;
    a.h[0] = *(const v8bf*)(arow + kb + hi * 8);
    a.h[1] = *(const v8bf*)(arow + kb + 16 + hi * 8);
    b.h[0] = *(const v8bf*)(bcol + kb + hi * 8);
    b.h[1] = *(const v8bf*)(bcol + kb + 16 + hi * 8);
    acc = __builtin_amdgcn_wmma_f32_16x16x32_bf16(false, a.v, false, b.v,
                                                  (short)0, acc, false, false);
  }
  return acc;
}

// -------- 1) structural projection: h[t,n,:] = x[t,n,:] @ Wflat (128x128) ----
__global__ __launch_bounds__(256) void k_struct_gemm(
    const float* __restrict__ x, long long sT, long long sN,
    const float* __restrict__ Ws,        // [SH,D,FH] for this layer
    float* __restrict__ h) {             // [T,N,D]
  __shared__ __bf16 sA[ROWS_][136];
  __shared__ __bf16 sWt[128][136];       // transposed weight: sWt[j][d]
  const int t   = blockIdx.y;
  const int m0  = blockIdx.x * ROWS_;
  const int tid = threadIdx.x;

  __builtin_prefetch(x + (long long)t * sT + (long long)m0 * sN, 0, 0);

  for (int i = tid; i < ROWS_ * 128; i += 256) {
    int r = i >> 7, c = i & 127;
    sA[r][c] = f2bf(x[(long long)t * sT + (long long)(m0 + r) * sN + c]);
  }
  // Wflat[d][j], j = head*16+f  ->  Ws[head*D*FH + d*FH + f]; stored transposed
  for (int i = tid; i < 128 * 128; i += 256) {
    int d = i >> 7, j = i & 127;
    sWt[j][d] = f2bf(Ws[(j >> 4) * (D_ * FH_) + d * FH_ + (j & 15)]);
  }
  __syncthreads();

  const int wave  = tid >> 5;
  const int lane  = tid & 31;
  const int colg  = wave * 16 + (lane & 15);
  const int rbase = (lane >> 4) * 8;
  float* dst = h + (long long)t * N_ * D_ + (long long)m0 * D_;
  for (int rt = 0; rt < ROWS_ / 16; rt++) {
    v8f acc = {};
    acc = wmma_tile(&sA[rt * 16][0], 136, &sWt[0][0], 136, wave * 16, lane, acc);
#pragma unroll
    for (int r = 0; r < 8; r++)
      dst[(long long)(rt * 16 + rbase + r) * D_ + colg] = acc[r];
  }
}

// -------- 2) per-head attention coefficients ps/pd ---------------------------
__global__ __launch_bounds__(256) void k_attn_coeff(
    const float* __restrict__ h, const float* __restrict__ as_,  // [SH,2*FH]
    float* __restrict__ ps, float* __restrict__ pd) {            // [T,SH,N]
  int idx = blockIdx.x * blockDim.x + threadIdx.x;
  if (idx >= T_ * N_) return;
  int t = idx / N_, n = idx - t * N_;
  const float* hp = h + (long long)t * N_ * D_ + (long long)n * D_;
#pragma unroll
  for (int hh = 0; hh < SH_; hh++) {
    float s1 = 0.f, s2 = 0.f;
#pragma unroll
    for (int f = 0; f < FH_; f++) {
      float v = hp[hh * FH_ + f];
      s1 += v * as_[hh * 2 * FH_ + f];
      s2 += v * as_[hh * 2 * FH_ + FH_ + f];
    }
    ps[((long long)t * SH_ + hh) * N_ + n] = s1;
    pd[((long long)t * SH_ + hh) * N_ + n] = s2;
  }
}

// -------- 3) edge gather / exp / scatter-add (L2 float atomics) --------------
// one thread per (t, edge, head): 8 consecutive threads cover one edge's
// 128 contiguous message floats; h[dst] read as float4s.
__global__ __launch_bounds__(256) void k_edge_scatter(
    const int* __restrict__ eidx, const float* __restrict__ evals,
    const float* __restrict__ h, const float* __restrict__ ps,
    const float* __restrict__ pd,
    float* __restrict__ rowsum,   // [T,N,SH]
    float* __restrict__ agg) {    // [T,N,D]
  long long idx = (long long)blockIdx.x * 256 + threadIdx.x;
  if (idx >= (long long)T_ * E_ * SH_) return;
  const int hh = (int)(idx & (SH_ - 1));
  long long te = idx >> 3;
  const int t = (int)(te / E_);
  const int e = (int)(te - (long long)t * E_);
  const int src = eidx[(long long)t * 2 * E_ + e];
  const int dst = eidx[(long long)t * 2 * E_ + E_ + e];
  const float w = evals[(long long)t * E_ + e];

  float s = w * (ps[((long long)t * SH_ + hh) * N_ + src] +
                 pd[((long long)t * SH_ + hh) * N_ + dst]);
  s = (s > 0.f) ? s : 0.2f * s;          // leaky_relu(0.2)
  const float ex = __expf(s);
  atomicAdd(&rowsum[((long long)t * N_ + src) * SH_ + hh], ex);

  const float4* hv = (const float4*)(h + (long long)t * N_ * D_ +
                                     (long long)dst * D_ + hh * FH_);
  float* ag = agg + (long long)t * N_ * D_ + (long long)src * D_ + hh * FH_;
#pragma unroll
  for (int q = 0; q < 4; q++) {
    float4 hx = hv[q];
    atomicAdd(&ag[q * 4 + 0], ex * hx.x);
    atomicAdd(&ag[q * 4 + 1], ex * hx.y);
    atomicAdd(&ag[q * 4 + 2], ex * hx.z);
    atomicAdd(&ag[q * 4 + 3], ex * hx.w);
  }
}

// -------- 4) sparse-softmax normalize + ELU (in place, float4) ---------------
__global__ __launch_bounds__(256) void k_gat_norm(
    const float* __restrict__ rowsum, float* __restrict__ agg) {
  long long i4 = (long long)blockIdx.x * 256 + threadIdx.x;
  if (i4 >= (long long)T_ * N_ * D_ / 4) return;
  long long i = i4 * 4;
  int d = (int)(i & (D_ - 1));
  long long tn = i >> 7;                        // t*N + n
  float inv = 1.f / rowsum[tn * SH_ + (d >> 4)];
  float4 v = ((float4*)agg)[i4];
  v.x *= inv; v.y *= inv; v.z *= inv; v.w *= inv;
  v.x = (v.x > 0.f) ? v.x : (__expf(v.x) - 1.f);
  v.y = (v.y > 0.f) ? v.y : (__expf(v.y) - 1.f);
  v.z = (v.z > 0.f) ? v.z : (__expf(v.z) - 1.f);
  v.w = (v.w > 0.f) ? v.w : (__expf(v.w) - 1.f);
  ((float4*)agg)[i4] = v;
}

// -------- 5) fused temporal attention (NODES_ nodes / workgroup) -------------
__global__ __launch_bounds__(256) void k_temporal(
    const float* __restrict__ gat,                         // [T,N,D]
    const float* __restrict__ Wq, const float* __restrict__ Wk,
    const float* __restrict__ Wv, const float* __restrict__ Wp,  // [T,D]
    float* __restrict__ out) {                             // [N,T,D]
  __shared__ __bf16 sA[NODES_ * 8][136];   // ti as bf16 (WMMA A operand)
  __shared__ __bf16 sWt[128][136];         // transposed projection matrix
  __shared__ float  sTi[NODES_ * 8][128];  // f32 ti for residual
  __shared__ float  sP[3][NODES_ * 8][128];// q,k,v tiles
  const int n0  = blockIdx.x * NODES_;
  const int tid = threadIdx.x;

  for (int i = tid; i < NODES_ * 8 * 128; i += 256) {
    int r = i >> 7, d = i & 127;
    int ln = r >> 3, tt = r & 7;
    float v = gat[(long long)tt * N_ * D_ + (long long)(n0 + ln) * D_ + d]
              + Wp[tt * D_ + d];
    sTi[r][d] = v;
    sA[r][d]  = f2bf(v);
  }
  __syncthreads();

  const int wave  = tid >> 5, lane = tid & 31;
  const int colg  = wave * 16 + (lane & 15);
  const int rbase = (lane >> 4) * 8;
  const float* Wm[3] = {Wq, Wk, Wv};
  for (int p = 0; p < 3; p++) {
    const float* W = Wm[p];
    for (int i = tid; i < 128 * 128; i += 256) {
      int d = i >> 7, j = i & 127;
      sWt[j][d] = f2bf(W[d * D_ + j]);   // einsum 'ntd,df' -> B[d][j] = W[d*D+j]
    }
    __syncthreads();
    for (int rt = 0; rt < NODES_ * 8 / 16; rt++) {
      v8f acc = {};
      acc = wmma_tile(&sA[rt * 16][0], 136, &sWt[0][0], 136, wave * 16, lane, acc);
#pragma unroll
      for (int r = 0; r < 8; r++)
        sP[p][rt * 16 + rbase + r][colg] = acc[r];
    }
    __syncthreads();
  }

  // causal softmax over T=8; 256 threads = node(4) x head(8) x tq(8), dh=16
  const int node = tid >> 6;
  const int head = (tid >> 3) & 7;
  const int tq   = tid & 7;
  const int r    = node * 8 + tq;
  float qv[16];
#pragma unroll
  for (int f = 0; f < 16; f++) qv[f] = sP[0][r][head * 16 + f];
  float sc[8];
  float mx = -1e30f;
  for (int tk = 0; tk <= tq; tk++) {
    float s = 0.f;
#pragma unroll
    for (int f = 0; f < 16; f++) s += qv[f] * sP[1][node * 8 + tk][head * 16 + f];
    s *= 0.25f;                          // 1/sqrt(dh), dh=16
    sc[tk] = s;
    mx = fmaxf(mx, s);
  }
  float den = 0.f;
  for (int tk = 0; tk <= tq; tk++) { float e = __expf(sc[tk] - mx); sc[tk] = e; den += e; }
  const float inv = 1.f / den;
  const long long ob = (long long)(n0 + node) * T_ * D_ + (long long)tq * D_ + head * 16;
#pragma unroll
  for (int f = 0; f < 16; f++) {
    float o = 0.f;
    for (int tk = 0; tk <= tq; tk++) o += sc[tk] * sP[2][node * 8 + tk][head * 16 + f];
    out[ob + f] = o * inv + sTi[r][head * 16 + f];
  }
}

extern "C" void kernel_launch(void* const* d_in, const int* in_sizes, int n_in,
                              void* d_out, int out_size, void* d_ws, size_t ws_size,
                              hipStream_t stream) {
  (void)in_sizes; (void)n_in; (void)out_size; (void)ws_size;
  const float* features  = (const float*)d_in[0];
  const int*   edge_idx  = (const int*)  d_in[1];
  const float* edge_vals = (const float*)d_in[2];
  const float* W_s = (const float*)d_in[3];
  const float* a_s = (const float*)d_in[4];
  const float* Wq  = (const float*)d_in[5];
  const float* Wk  = (const float*)d_in[6];
  const float* Wv  = (const float*)d_in[7];
  const float* Wp  = (const float*)d_in[8];
  float* out = (float*)d_out;

  float* ws = (float*)d_ws;
  const size_t big = (size_t)T_ * N_ * D_;     // 51.2M floats
  const size_t sml = (size_t)T_ * SH_ * N_;    // 3.2M floats
  float* hbuf   = ws;                // [T,N,D] structural projection
  float* agg    = ws + big;          // [T,N,D] message agg -> GAT output
  float* inter  = ws + 2 * big;      // [N,T,D] layer-0 temporal output
  float* ps     = ws + 3 * big;      // [T,SH,N]
  float* pd     = ps + sml;          // [T,SH,N]
  float* rowsum = pd + sml;          // [T,N,SH]

  for (int ell = 0; ell < L_; ell++) {
    const float* x; long long sT, sN;
    if (ell == 0) { x = features; sT = (long long)N_ * D_; sN = D_; }
    else          { x = inter;    sT = D_;                 sN = (long long)T_ * D_; }
    float* lout = (ell == L_ - 1) ? out : inter;

    hipMemsetAsync(agg,    0, big * sizeof(float), stream);
    hipMemsetAsync(rowsum, 0, sml * sizeof(float), stream);

    k_struct_gemm<<<dim3(N_ / ROWS_, T_), 256, 0, stream>>>(
        x, sT, sN, W_s + (size_t)ell * SH_ * D_ * FH_, hbuf);
    k_attn_coeff<<<(T_ * N_ + 255) / 256, 256, 0, stream>>>(
        hbuf, a_s + (size_t)ell * SH_ * 2 * FH_, ps, pd);
    k_edge_scatter<<<(int)(((long long)T_ * E_ * SH_ + 255) / 256), 256, 0, stream>>>(
        edge_idx, edge_vals, hbuf, ps, pd, rowsum, agg);
    k_gat_norm<<<(int)(((long long)T_ * N_ * D_ / 4 + 255) / 256), 256, 0, stream>>>(
        rowsum, agg);
    k_temporal<<<N_ / NODES_, 256, 0, stream>>>(
        agg, Wq + (size_t)ell * D_ * D_, Wk + (size_t)ell * D_ * D_,
        Wv + (size_t)ell * D_ * D_, Wp + (size_t)ell * T_ * D_, lout);
  }
}
